// SurfacePrediction_62792421867848
// MI455X (gfx1250) — matrix operand order repaired and checked by reading, not compiled, not run on hardware
//
#include <hip/hip_runtime.h>
#include <hip/hip_bf16.h>

typedef __attribute__((ext_vector_type(16))) _Float16 v16h;
typedef __attribute__((ext_vector_type(8)))  _Float16 v8h;
typedef __attribute__((ext_vector_type(8)))  float    v8f;

#define SCN_EPS 1e-4f

// ---------------------------------------------------------------------------
// p1: submanifold conv, Cin=2 -> Cout=16, 27 offsets. Cheap scalar kernel.
// ---------------------------------------------------------------------------
__global__ void p1_kernel(const float* __restrict__ feats,
                          const int*   __restrict__ nbr,   // [27, N]
                          const float* __restrict__ w,     // [27, 2, 16]
                          float*       __restrict__ out,   // [N, 16]
                          int N) {
    int i = blockIdx.x * blockDim.x + threadIdx.x;
    if (i >= N) return;
    float acc[16];
#pragma unroll
    for (int n = 0; n < 16; ++n) acc[n] = 0.f;
#pragma unroll 1
    for (int k = 0; k < 27; ++k) {
        int idx = nbr[(size_t)k * N + i];
        if (idx < N) {
            float f0 = feats[(size_t)idx * 2 + 0];
            float f1 = feats[(size_t)idx * 2 + 1];
            const float* wk = w + k * 32;
#pragma unroll
            for (int n = 0; n < 16; ++n)
                acc[n] += f0 * wk[n] + f1 * wk[16 + n];
        }
    }
#pragma unroll
    for (int n = 0; n < 16; ++n) out[(size_t)i * 16 + n] = acc[n];
}

// ---------------------------------------------------------------------------
// Per-channel sum / sum-of-squares for C=16 tensors (SCN batchnorm stats).
// stats[0..15] = sum, stats[16..31] = sumsq. Must be zeroed before launch.
// ---------------------------------------------------------------------------
__global__ void bn_stats16(const float* __restrict__ x, int N,
                           float* __restrict__ stats) {
    __shared__ float ss[256];
    __shared__ float sq[256];
    int tid = threadIdx.x;
    int c = tid & 15;
    float s = 0.f, q = 0.f;
    size_t total = (size_t)N * 16;
    for (size_t e = (size_t)blockIdx.x * 256 + tid; e < total;
         e += (size_t)gridDim.x * 256) {
        float v = x[e];
        s += v; q += v * v;
    }
    ss[tid] = s; sq[tid] = q;
    __syncthreads();
    for (int off = 128; off >= 16; off >>= 1) {
        if (tid < off) { ss[tid] += ss[tid + off]; sq[tid] += sq[tid + off]; }
        __syncthreads();
    }
    if (tid < 16) {
        atomicAdd(&stats[c], ss[tid]);
        atomicAdd(&stats[16 + c], sq[tid]);
    }
}

// ---------------------------------------------------------------------------
// BN+ReLU, write f16 activation table with zeroed sentinel row N.
// ---------------------------------------------------------------------------
__global__ void bnrelu16_f16(const float* __restrict__ x,
                             const float* __restrict__ stats,
                             const float* __restrict__ g,
                             const float* __restrict__ b,
                             _Float16* __restrict__ out,  // [(N+1), 16]
                             int N, float invN) {
    size_t e = (size_t)blockIdx.x * blockDim.x + threadIdx.x;
    size_t total = (size_t)(N + 1) * 16;
    if (e >= total) return;
    if (e >= (size_t)N * 16) { out[e] = (_Float16)0.f; return; }
    int c = (int)(e & 15);
    float mean = stats[c] * invN;
    float var  = stats[16 + c] * invN - mean * mean;
    float y = (x[e] - mean) * __frsqrt_rn(var + SCN_EPS) * g[c] + b[c];
    out[e] = (_Float16)(y > 0.f ? y : 0.f);
}

// ---------------------------------------------------------------------------
// WMMA rulebook sparse conv: out[s, :] = res[s, :] + sum_k xf16[nbr[k,s], :] @ W[k]
// xf16: [(Nin+1), 16] f16, row Nin zeroed (sentinel). W: [K,16,16] f32.
// One wave per 16-site tile; offsets paired into the K=32 dimension of
// v_wmma_f32_16x16x32_f16. K is compile-time -> fully unrolled WMMA chain.
// ---------------------------------------------------------------------------
template <int K>
__global__ void sconv_wmma(const _Float16* __restrict__ xf16,
                           const int*      __restrict__ nbr,   // [K, Nout]
                           const float*    __restrict__ W,     // [K, 16, 16]
                           const float*    __restrict__ res,   // [Nout,16] or null
                           float*          __restrict__ out,   // [Nout,16]
                           int Nin, int Nout) {
    constexpr int KP = (K + 1) & ~1;         // pad to even #offsets
    // Stage weights to LDS transposed to [k][n][c] in f16 (pad slot zeroed).
    __shared__ __align__(32) _Float16 Wl[KP * 256];
    for (int e = threadIdx.x; e < KP * 256; e += blockDim.x) {
        int k = e >> 8, n = (e >> 4) & 15, c = e & 15;
        float v = (k < K) ? W[((size_t)k * 16 + c) * 16 + n] : 0.f;
        Wl[e] = (_Float16)v;
    }
    __syncthreads();

    int lane = threadIdx.x & 31;
    int wid  = threadIdx.x >> 5;
    int tile = blockIdx.x * 4 + wid;
    if (tile * 16 >= Nout) return;           // uniform per wave

    int  m   = lane & 15;                    // row M for A, column N for B/D
    bool hi  = lane >= 16;
    int  off = hi ? 8 : 0;                   // channel half gathered by this lane
    int  s   = tile * 16 + m;
    bool sok = (s < Nout);

    // Residual rides in the WMMA C accumulator.
    // C/D layout: lane l, VGPR j -> element [j + (hi?8:0)][l%16].
    v8f acc;
    if (res) {
#pragma unroll
        for (int j = 0; j < 8; ++j) {
            int ss = tile * 16 + j + (hi ? 8 : 0);
            acc[j] = (ss < Nout) ? res[(size_t)ss * 16 + m] : 0.f;
        }
    } else {
#pragma unroll
        for (int j = 0; j < 8; ++j) acc[j] = 0.f;
    }

#pragma unroll
    for (int kp = 0; kp < KP; kp += 2) {
        const int k0 = kp, k1 = kp + 1;
        int idx0 = (sok)           ? nbr[(size_t)k0 * Nout + s] : Nin;
        int idx1 = (sok && k1 < K) ? nbr[(size_t)k1 * Nout + s] : Nin;

        // A: 16x32 f16. lane<16: K 0-7 (k0 ch0-7) + K 16-23 (k1 ch0-7);
        //               lane>=16: K 8-15 (k0 ch8-15) + K 24-31 (k1 ch8-15).
        v8h a0 = *reinterpret_cast<const v8h*>(xf16 + (size_t)idx0 * 16 + off);
        v8h a1 = *reinterpret_cast<const v8h*>(xf16 + (size_t)idx1 * 16 + off);
        v16h A;
#pragma unroll
        for (int i = 0; i < 8; ++i) { A[i] = a0[i]; A[8 + i] = a1[i]; }

        // B: 32x16 f16. lane<16: column m, K 0-7 = W[k0][0..7][m], K 16-23 = W[k1][0..7][m];
        //               lane>=16: column m, K 8-15 = W[k0][8..15][m], K 24-31 = W[k1][8..15][m].
        v8h b0 = *reinterpret_cast<const v8h*>(&Wl[((k0 * 16 + m) << 4) + off]);
        v8h b1 = *reinterpret_cast<const v8h*>(&Wl[((k1 * 16 + m) << 4) + off]);
        v16h B;
#pragma unroll
        for (int i = 0; i < 8; ++i) { B[i] = b0[i]; B[8 + i] = b1[i]; }

        acc = __builtin_amdgcn_wmma_f32_16x16x32_f16(
            false, A, false, B, (short)0, acc, false, false);
    }

#pragma unroll
    for (int j = 0; j < 8; ++j) {
        int ss = tile * 16 + j + (hi ? 8 : 0);
        if (ss < Nout) out[(size_t)ss * 16 + m] = acc[j];
    }
}

// ---------------------------------------------------------------------------
// Stats for the virtual 48-channel join j0[i] = [r0[i], r1[p01[i]], r2[p12[p01[i]]]]
// stats[0..47]=sum, stats[48..95]=sumsq. Must be zeroed before launch.
// ---------------------------------------------------------------------------
__global__ void join_stats48(const float* __restrict__ r0,
                             const float* __restrict__ r1,
                             const float* __restrict__ r2,
                             const int* __restrict__ p01,
                             const int* __restrict__ p12,
                             int N, float* __restrict__ stats) {
    __shared__ float ss[192];
    __shared__ float sq[192];
    int tid = threadIdx.x;
    int c = tid % 48, grp = tid / 48;        // 4 site-groups per block
    float s = 0.f, q = 0.f;
    for (int i = blockIdx.x * 4 + grp; i < N; i += gridDim.x * 4) {
        float v;
        if (c < 16)      v = r0[(size_t)i * 16 + c];
        else if (c < 32) v = r1[(size_t)p01[i] * 16 + (c - 16)];
        else             v = r2[(size_t)p12[p01[i]] * 16 + (c - 32)];
        s += v; q += v * v;
    }
    ss[tid] = s; sq[tid] = q;
    __syncthreads();
    if (tid < 96) { ss[tid] += ss[tid + 96]; sq[tid] += sq[tid + 96]; }
    __syncthreads();
    if (tid < 48) {
        ss[tid] += ss[tid + 48]; sq[tid] += sq[tid + 48];
        atomicAdd(&stats[c], ss[tid]);
        atomicAdd(&stats[48 + c], sq[tid]);
    }
}

// ---------------------------------------------------------------------------
// Final: BN+ReLU over 48 joined channels, dot with w_sdf, add b_sdf.
// ---------------------------------------------------------------------------
__global__ void final_kernel(const float* __restrict__ r0,
                             const float* __restrict__ r1,
                             const float* __restrict__ r2,
                             const int* __restrict__ p01,
                             const int* __restrict__ p12,
                             const float* __restrict__ stats,
                             const float* __restrict__ g,
                             const float* __restrict__ b,
                             const float* __restrict__ wsdf,
                             const float* __restrict__ bsdf,
                             float* __restrict__ out,
                             int N, float invN) {
    int i = blockIdx.x * blockDim.x + threadIdx.x;
    if (i >= N) return;
    int i1 = p01[i];
    int i2 = p12[i1];
    float acc = bsdf[0];
#pragma unroll 4
    for (int c = 0; c < 48; ++c) {
        float v = (c < 16) ? r0[(size_t)i * 16 + c]
                : (c < 32) ? r1[(size_t)i1 * 16 + (c - 16)]
                           : r2[(size_t)i2 * 16 + (c - 32)];
        float mean = stats[c] * invN;
        float var  = stats[48 + c] * invN - mean * mean;
        float y = (v - mean) * __frsqrt_rn(var + SCN_EPS) * g[c] + b[c];
        acc += (y > 0.f ? y : 0.f) * wsdf[c];
    }
    out[i] = acc;
}

// ---------------------------------------------------------------------------
// Host orchestration
// ---------------------------------------------------------------------------
extern "C" void kernel_launch(void* const* d_in, const int* in_sizes, int n_in,
                              void* d_out, int out_size, void* d_ws, size_t ws_size,
                              hipStream_t stream) {
    const float* feats     = (const float*)d_in[0];
    const float* w_p1      = (const float*)d_in[1];
    const float* blk_bn1_g = (const float*)d_in[2];
    const float* blk_bn1_b = (const float*)d_in[3];
    const float* blk_w1    = (const float*)d_in[4];   // [3,27,16,16]
    const float* blk_bn2_g = (const float*)d_in[5];
    const float* blk_bn2_b = (const float*)d_in[6];
    const float* blk_w2    = (const float*)d_in[7];   // [3,27,16,16]
    const float* down_bn_g = (const float*)d_in[8];
    const float* down_bn_b = (const float*)d_in[9];
    const float* down_w    = (const float*)d_in[10];  // [2,8,16,16]
    const float* bn3_g     = (const float*)d_in[11];
    const float* bn3_b     = (const float*)d_in[12];
    const float* w_sdf     = (const float*)d_in[13];
    const float* b_sdf     = (const float*)d_in[14];
    const int*   nbr0      = (const int*)d_in[15];
    const int*   nbr1      = (const int*)d_in[16];
    const int*   nbr2      = (const int*)d_in[17];
    const int*   down01    = (const int*)d_in[18];
    const int*   down12    = (const int*)d_in[19];
    const int*   p01       = (const int*)d_in[20];
    const int*   p12       = (const int*)d_in[21];

    const int N0 = in_sizes[0] / 2;
    const int N1 = in_sizes[16] / 27;
    const int N2 = in_sizes[17] / 27;

    // Workspace carve-out (units: floats, 256B-aligned chunks).
    auto au = [](size_t v) { return (v + 63) & ~(size_t)63; };
    float* ws = (float*)d_ws;
    size_t o = 0;
    float* X  = ws + o; o += au((size_t)N0 * 16);   // level input / reused
    float* H  = ws + o; o += au((size_t)N0 * 16);   // intra-block temp
    float* R0 = ws + o; o += au((size_t)N0 * 16);
    float* R1 = ws + o; o += au((size_t)N1 * 16);
    float* R2 = ws + o; o += au((size_t)N2 * 16);
    float* ST = ws + o; o += au(96);                // BN stats (sum, sumsq)
    _Float16* A = (_Float16*)(ws + o);              // f16 activations, padded row
    o += au(((size_t)N0 + 1) * 8);
    (void)ws_size; (void)n_in; (void)out_size;

    auto stats16 = [&](const float* x, int N) {
        hipMemsetAsync(ST, 0, 96 * sizeof(float), stream);
        int blocks = (int)(((size_t)N * 16 + 255) / 256);
        if (blocks > 1024) blocks = 1024;
        if (blocks < 1) blocks = 1;
        bn_stats16<<<blocks, 256, 0, stream>>>(x, N, ST);
    };
    auto bnrelu = [&](const float* x, const float* g, const float* b, int N) {
        size_t total = (size_t)(N + 1) * 16;
        bnrelu16_f16<<<(int)((total + 255) / 256), 256, 0, stream>>>(
            x, ST, g, b, A, N, 1.0f / (float)N);
    };
    auto conv27 = [&](const int* nbr, const float* W, const float* res,
                      float* out, int Nin, int Nout) {
        int blocks = (((Nout + 15) / 16) + 3) / 4;
        if (blocks < 1) blocks = 1;
        sconv_wmma<27><<<blocks, 128, 0, stream>>>(A, nbr, W, res, out, Nin, Nout);
    };
    auto conv8 = [&](const int* nbr, const float* W, const float* res,
                     float* out, int Nin, int Nout) {
        int blocks = (((Nout + 15) / 16) + 3) / 4;
        if (blocks < 1) blocks = 1;
        sconv_wmma<8><<<blocks, 128, 0, stream>>>(A, nbr, W, res, out, Nin, Nout);
    };
    auto resblock = [&](float* Xb, float* Rb, const int* nbr, int N, int i) {
        stats16(Xb, N);
        bnrelu(Xb, blk_bn1_g + i * 16, blk_bn1_b + i * 16, N);
        conv27(nbr, blk_w1 + (size_t)i * 27 * 256, nullptr, H, N, N);
        stats16(H, N);
        bnrelu(H, blk_bn2_g + i * 16, blk_bn2_b + i * 16, N);
        conv27(nbr, blk_w2 + (size_t)i * 27 * 256, Xb, Rb, N, N);
    };

    // p1: feats -> X  (N0 x 16)
    p1_kernel<<<(N0 + 255) / 256, 256, 0, stream>>>(feats, nbr0, w_p1, X, N0);

    // Level 0 block -> R0
    resblock(X, R0, nbr0, N0, 0);
    // Downsample 0: R0 -> X (N1 x 16)
    stats16(R0, N0);
    bnrelu(R0, down_bn_g + 0, down_bn_b + 0, N0);
    conv8(down01, down_w + 0, nullptr, X, N0, N1);
    // Level 1 block -> R1
    resblock(X, R1, nbr1, N1, 1);
    // Downsample 1: R1 -> X (N2 x 16)
    stats16(R1, N1);
    bnrelu(R1, down_bn_g + 16, down_bn_b + 16, N1);
    conv8(down12, down_w + 8 * 256, nullptr, X, N1, N2);
    // Level 2 block -> R2
    resblock(X, R2, nbr2, N2, 2);

    // Join stats over virtual 48-channel concat, then fused BN+ReLU+linear.
    hipMemsetAsync(ST, 0, 96 * sizeof(float), stream);
    {
        int blocks = (N0 + 3) / 4;
        if (blocks > 2048) blocks = 2048;
        join_stats48<<<blocks, 192, 0, stream>>>(R0, R1, R2, p01, p12, N0, ST);
    }
    final_kernel<<<(N0 + 255) / 256, 256, 0, stream>>>(
        R0, R1, R2, p01, p12, ST, bn3_g, bn3_b, w_sdf, b_sdf,
        (float*)d_out, N0, 1.0f / (float)N0);
}
